// CrossGeometricStructureEmbedding_23390391894722
// MI455X (gfx1250) — matrix operand order repaired and checked by reading, not compiled, not run on hardware
//
#include <hip/hip_runtime.h>

// ---------------------------------------------------------------------------
// CrossGeometricStructureEmbedding for MI455X (gfx1250, wave32, WMMA)
//   B=1, N=4096, K=64, H=256
//   Two (N*K,256)x(256,256) GEMMs via v_wmma_f32_16x16x32_f16, max-over-K
//   fused into the epilogue. v2: 2 column tiles per wave -> each A fragment
//   (LDS) feeds 2 WMMAs, halving DS traffic (LDS-roofline fix).
// ---------------------------------------------------------------------------

typedef __attribute__((ext_vector_type(16))) _Float16 v16h;
typedef __attribute__((ext_vector_type(8)))  _Float16 v8h;
typedef __attribute__((ext_vector_type(8)))  float    v8f;

#define HID     256
#define KANCH   64
#define SIGMA_D 0.2f
#define FACTOR_A 3.8197186342054885f          // 180 / (15 * pi)
#define DIVLOG2  0.10381025296523007f         // log2(10000) / 128

// ---- prep: convert Wa, Wd (f32 row-major HxH) to f16 into workspace --------
__global__ void prep_w_f16(const float* __restrict__ Wa,
                           const float* __restrict__ Wd,
                           _Float16* __restrict__ WaH,
                           _Float16* __restrict__ WdH) {
  int i = blockIdx.x * blockDim.x + threadIdx.x;
  if (i < HID * HID) {
    WaH[i] = (_Float16)Wa[i];
    WdH[i] = (_Float16)Wd[i];
  }
}

__device__ __forceinline__ v16h cat8(v8h lo, v8h hi) {
  return __builtin_shufflevector(lo, hi,
           0, 1, 2, 3, 4, 5, 6, 7, 8, 9, 10, 11, 12, 13, 14, 15);
}

// ---- main kernel: one workgroup (256 thr = 8 waves) per point n ------------
__global__ __launch_bounds__(256, 2)
void cgse_kernel(const float* __restrict__ points,    // (N,3)
                 const float* __restrict__ anchors,   // (K,3)
                 const _Float16* __restrict__ WaH,    // (H,H) f16
                 const float* __restrict__ ba,        // (H,)
                 const _Float16* __restrict__ WdH,    // (H,H) f16
                 const float* __restrict__ bd,        // (H,)
                 float* __restrict__ out) {           // (N,H)
  __shared__ float sIdxD[KANCH];
  __shared__ float sIdxA[KANCH];
  __shared__ _Float16 emb[KANCH * HID];               // 32 KB, reused a -> d

  const int n    = blockIdx.x;
  const int t    = threadIdx.x;
  const int lane = t & 31;
  const int wv   = t >> 5;                            // 0..7, owns 2 col tiles

  // ---- phase 0: distance / angle indices (one lane per anchor k) ----------
  if (t < KANCH) {
    const float px = points[n * 3 + 0];
    const float py = points[n * 3 + 1];
    const float pz = points[n * 3 + 2];
    const int   k2 = (t + 1) & (KANCH - 1);           // roll(-1) along K
    const float rx = px - anchors[t * 3 + 0];
    const float ry = py - anchors[t * 3 + 1];
    const float rz = pz - anchors[t * 3 + 2];
    const float qx = px - anchors[k2 * 3 + 0];
    const float qy = py - anchors[k2 * 3 + 1];
    const float qz = pz - anchors[k2 * 3 + 2];
    sIdxD[t] = sqrtf(rx * rx + ry * ry + rz * rz) * (1.0f / SIGMA_D);
    const float cx = ry * qz - rz * qy;
    const float cy = rz * qx - rx * qz;
    const float cz = rx * qy - ry * qx;
    const float sv = sqrtf(cx * cx + cy * cy + cz * cz);
    const float cv = rx * qx + ry * qy + rz * qz;
    sIdxA[t] = atan2f(sv, cv) * FACTOR_A;
  }
  __syncthreads();

  const int g = lane >> 4;                            // half-wave group
  const int m = lane & 15;
  float colres[2][2];                                 // [phase][tile]

  #pragma unroll
  for (int p = 0; p < 2; ++p) {                       // p=0: angle/Wa, p=1: dist/Wd
    const float*    xi = (p == 0) ? sIdxA : sIdxD;
    const _Float16* W  = (p == 0) ? WaH : WdH;

    // ---- build 64x256 f16 embedding tile in LDS (packed sin|cos words) ----
    unsigned int* emb32 = (unsigned int*)emb;
    #pragma unroll
    for (int it = 0; it < 32; ++it) {
      const int idx = t + it * 256;                   // 0..8191
      const int k   = idx >> 7;
      const int i   = idx & 127;
      const float w = xi[k] * exp2f(-(float)i * DIVLOG2);
      union { _Float16 h[2]; unsigned int u; } pk;
      pk.h[0] = (_Float16)__sinf(w);
      pk.h[1] = (_Float16)__cosf(w);
      emb32[k * 128 + i] = pk.u;
    }
    __syncthreads();

    // ---- GEMM: wave `wv` owns output columns [wv*32, wv*32+32) ------------
    // B fragment (32x16, K contiguous per half-wave): column o, lane group g
    // selects K range [g*16, g*16+16) within each k-step.
    const int o0 = wv * 32 + m;                       // col tile 0
    const int o1 = o0 + 16;                           // col tile 1
    const v8h* wrow0 = (const v8h*)(W + o0 * HID);
    const v8h* wrow1 = (const v8h*)(W + o1 * HID);

    v8f acc[8];                                       // [r][tile], 4x2 block
    #pragma unroll
    for (int q = 0; q < 8; ++q)
      acc[q] = (v8f){0.f, 0.f, 0.f, 0.f, 0.f, 0.f, 0.f, 0.f};

    #pragma unroll
    for (int s = 0; s < 8; ++s) {
      const v16h b0 = cat8(wrow0[4 * s + 2 * g], wrow0[4 * s + 2 * g + 1]);
      const v16h b1 = cat8(wrow1[4 * s + 2 * g], wrow1[4 * s + 2 * g + 1]);
      #pragma unroll
      for (int r = 0; r < 4; ++r) {
        // A fragment (16x32): lane row r*16+m, K contiguous [g*16, g*16+16)
        const v8h* arow = (const v8h*)(emb + (r * 16 + m) * HID);
        const v16h a = cat8(arow[4 * s + 2 * g], arow[4 * s + 2 * g + 1]);
        acc[2 * r + 0] = __builtin_amdgcn_wmma_f32_16x16x32_f16(
            false, a, false, b0, (short)0, acc[2 * r + 0], false, false);
        acc[2 * r + 1] = __builtin_amdgcn_wmma_f32_16x16x32_f16(
            false, a, false, b1, (short)0, acc[2 * r + 1], false, false);
      }
    }

    // ---- epilogue: max over K rows ----------------------------------------
    float mx0 = -3.402823466e38f, mx1 = -3.402823466e38f;
    #pragma unroll
    for (int r = 0; r < 4; ++r) {
      #pragma unroll
      for (int v = 0; v < 8; ++v) {
        mx0 = fmaxf(mx0, acc[2 * r + 0][v]);
        mx1 = fmaxf(mx1, acc[2 * r + 1][v]);
      }
    }
    // lanes l and l+16 hold the two halves of the row-max for each column
    colres[p][0] = fmaxf(mx0, __shfl_xor(mx0, 16, 32));
    colres[p][1] = fmaxf(mx1, __shfl_xor(mx1, 16, 32));
    __syncthreads();                                  // before emb is rebuilt
  }

  if (lane < 16) {
    const int oA = wv * 32 + lane;
    const int oB = oA + 16;
    out[n * HID + oA] = colres[0][0] + colres[1][0] + ba[oA] + bd[oA];
    out[n * HID + oB] = colres[0][1] + colres[1][1] + ba[oB] + bd[oB];
  }
}

// ---------------------------------------------------------------------------
extern "C" void kernel_launch(void* const* d_in, const int* in_sizes, int n_in,
                              void* d_out, int out_size, void* d_ws, size_t ws_size,
                              hipStream_t stream) {
  const float* points  = (const float*)d_in[0];       // (1,N,3)
  const float* anchors = (const float*)d_in[1];       // (1,K,3)
  // d_in[2] cor_score: unused by the reference
  const float* Wa = (const float*)d_in[3];            // (H,H)
  const float* ba = (const float*)d_in[4];            // (H,)
  const float* Wd = (const float*)d_in[5];            // (H,H)
  const float* bd = (const float*)d_in[6];            // (H,)
  float* out = (float*)d_out;                         // (1,N,H)

  const int N = in_sizes[0] / 3;

  _Float16* WaH = (_Float16*)d_ws;                    // 128 KB
  _Float16* WdH = WaH + HID * HID;                    // 128 KB

  prep_w_f16<<<(HID * HID + 255) / 256, 256, 0, stream>>>(Wa, Wd, WaH, WdH);
  cgse_kernel<<<N, 256, 0, stream>>>(points, anchors, WaH, ba, WdH, bd, out);
}
